// SeparabilityLoss_24739011625309
// MI455X (gfx1250) — compile-verified, matrix operand order checked by reading, not run on hardware
//
#include <hip/hip_runtime.h>
#include <hip/hip_bf16.h>

typedef __bf16 bf16_t;
typedef __attribute__((ext_vector_type(16))) __bf16 v16bf;
typedef __attribute__((ext_vector_type(8)))  __bf16 v8bf;
typedef __attribute__((ext_vector_type(8)))  float  v8f;

#define N_IMG   64
#define P_PATCH 256
#define D_DIM   64
#define LDS_STRIDE 72   // 64 + 8 pad: 36*row mod 64 distinct for 16 rows -> no bank conflicts

// ---------------- Kernel 0: L2-normalize rows, convert to bf16 ----------------
__global__ void __launch_bounds__(256)
normalize_bf16_kernel(const float* __restrict__ normal,
                      const float* __restrict__ defect,
                      bf16_t* __restrict__ a_out,
                      bf16_t* __restrict__ b_out) {
  // one wave32 per D=64 row (2 floats / lane); 8 waves per block
  int warp = (blockIdx.x * blockDim.x + threadIdx.x) >> 5;
  int lane = threadIdx.x & 31;
  const int rows_per_mat = N_IMG * P_PATCH;        // 16384
  if (warp >= 2 * rows_per_mat) return;
  const float* src; bf16_t* dst; int row;
  if (warp < rows_per_mat) { src = normal; dst = a_out; row = warp; }
  else                     { src = defect; dst = b_out; row = warp - rows_per_mat; }
  const float* r = src + (size_t)row * D_DIM;
  float x0 = r[lane], x1 = r[lane + 32];
  float ss = x0 * x0 + x1 * x1;
  #pragma unroll
  for (int off = 16; off > 0; off >>= 1) ss += __shfl_xor(ss, off, 32);
  float scale = 1.0f / (sqrtf(ss) + 1e-8f);        // x / (||x|| + eps)
  bf16_t* o = dst + (size_t)row * D_DIM;
  o[lane]      = (bf16_t)(x0 * scale);
  o[lane + 32] = (bf16_t)(x1 * scale);
}

// ---------------- Kernel 1: per-pair fused GEMM + max/mean ----------------
// One workgroup (512 threads = 16 waves) per (a-image, b-image) pair.
// Wave tiling: 4 row-groups x 4 column-groups; each wave computes a 64x64
// output block as 4 strips x 4 tiles, 8 WMMAs per B-fragment load.
__global__ void __launch_bounds__(512)
pair_sim_kernel(const bf16_t* __restrict__ a_mat,   // [64,256,64] normalized bf16
                const bf16_t* __restrict__ b_mat,   // [64,256,64] normalized bf16
                float* __restrict__ sim_out) {      // [64*64]
  __shared__ bf16_t lds_a[P_PATCH * LDS_STRIDE];
  __shared__ bf16_t lds_b[P_PATCH * LDS_STRIDE];
  __shared__ float partial[4 * P_PATCH];            // [colgroup][row] row-max partials
  __shared__ float acc;

  const int pair = blockIdx.x;
  const int ni = pair >> 6;
  const int mi = pair & 63;
  const bf16_t* ga = a_mat + (size_t)ni * P_PATCH * D_DIM;
  const bf16_t* gb = b_mat + (size_t)mi * P_PATCH * D_DIM;

  const int tid = threadIdx.x;
  if (tid == 0) acc = 0.0f;

  // Cooperative stage: 256x64 bf16 per matrix as 16B chunks (2048 chunks, 4/thread)
  #pragma unroll
  for (int c = 0; c < 4; ++c) {
    int chunk = tid + c * 512;
    int row = chunk >> 3;
    int col = (chunk & 7) << 3;
    *(v8bf*)&lds_a[row * LDS_STRIDE + col] = *(const v8bf*)&ga[row * D_DIM + col];
    *(v8bf*)&lds_b[row * LDS_STRIDE + col] = *(const v8bf*)&gb[row * D_DIM + col];
  }
  __syncthreads();

  const int wave = tid >> 5;
  const int lane = tid & 31;
  const int ln = lane & 15;
  const int hb = lane >> 4;           // wave half (K sub-block select)
  const int rowgrp = wave >> 2;       // 0..3 : 64-row block
  const int colgrp = wave & 3;        // 0..3 : 4 column tiles (64 b-patches)

  // A fragments: 4 strips x (K=0..31, K=32..63); two contiguous 16B runs each
  v16bf a[4][2];
  #pragma unroll
  for (int s = 0; s < 4; ++s) {
    const bf16_t* base = &lds_a[(rowgrp * 64 + s * 16 + ln) * LDS_STRIDE + hb * 8];
    *(v8bf*)&a[s][0]       = *(const v8bf*)(base + 0);
    *((v8bf*)&a[s][0] + 1) = *(const v8bf*)(base + 16);
    *(v8bf*)&a[s][1]       = *(const v8bf*)(base + 32);
    *((v8bf*)&a[s][1] + 1) = *(const v8bf*)(base + 48);
  }

  float rm[4][8];
  #pragma unroll
  for (int s = 0; s < 4; ++s)
    #pragma unroll
    for (int j = 0; j < 8; ++j) rm[s][j] = -3.0e38f;

  #pragma unroll
  for (int t = 0; t < 4; ++t) {       // 4 column tiles for this wave
    const int ct = colgrp * 4 + t;
    const bf16_t* bb = &lds_b[(ct * 16 + ln) * LDS_STRIDE + hb * 8];
    v16bf b0, b1;
    *(v8bf*)&b0       = *(const v8bf*)(bb + 0);
    *((v8bf*)&b0 + 1) = *(const v8bf*)(bb + 16);
    *(v8bf*)&b1       = *(const v8bf*)(bb + 32);
    *((v8bf*)&b1 + 1) = *(const v8bf*)(bb + 48);
    // 4 independent accumulation chains -> WMMA pipe stays busy, hazards hidden
    #pragma unroll
    for (int s = 0; s < 4; ++s) {
      v8f c = {};
      c = __builtin_amdgcn_wmma_f32_16x16x32_bf16(false, a[s][0], false, b0, (short)0, c, false, false);
      c = __builtin_amdgcn_wmma_f32_16x16x32_bf16(false, a[s][1], false, b1, (short)0, c, false, false);
      #pragma unroll
      for (int j = 0; j < 8; ++j) rm[s][j] = fmaxf(rm[s][j], c[j]);
    }
  }

  // max over the 16 N-lanes within each half (rows j in lanes 0-15, rows 8+j in 16-31)
  #pragma unroll
  for (int off = 8; off > 0; off >>= 1) {
    #pragma unroll
    for (int s = 0; s < 4; ++s)
      #pragma unroll
      for (int j = 0; j < 8; ++j) rm[s][j] = fmaxf(rm[s][j], __shfl_xor(rm[s][j], off, 32));
  }

  // lanes 0 and 16 publish this wave's per-row maxima over its 64 columns
  if (ln == 0) {
    #pragma unroll
    for (int s = 0; s < 4; ++s) {
      int row = rowgrp * 64 + s * 16 + hb * 8;
      #pragma unroll
      for (int j = 0; j < 8; ++j) partial[colgrp * P_PATCH + row + j] = rm[s][j];
    }
  }
  __syncthreads();

  // combine the 4 column-group maxima and take the mean over the 256 rows
  float contrib = 0.0f;
  if (tid < P_PATCH) {
    float m = partial[tid];
    #pragma unroll
    for (int cg = 1; cg < 4; ++cg) m = fmaxf(m, partial[cg * P_PATCH + tid]);
    contrib = m;
  }
  #pragma unroll
  for (int off = 16; off > 0; off >>= 1) contrib += __shfl_xor(contrib, off, 32);
  if (lane == 0) atomicAdd(&acc, contrib);
  __syncthreads();
  if (tid == 0) sim_out[pair] = acc * (1.0f / (float)P_PATCH);
}

// ---------------- Kernel 2: scalar statistics ----------------
__global__ void __launch_bounds__(256)
stats_kernel(const float* __restrict__ sim_nn,
             const float* __restrict__ sim_nd,
             float* __restrict__ out) {
  int tid = threadIdx.x;
  float snn = 0.f, snn2 = 0.f, snd = 0.f, snd2 = 0.f;
  for (int i = tid; i < 4096; i += 256) {
    float vnd = sim_nd[i];
    snd += vnd; snd2 += vnd * vnd;
    int r = i >> 6, c = i & 63;
    if (r != c) { float v = sim_nn[i]; snn += v; snn2 += v * v; }
  }
  #pragma unroll
  for (int off = 16; off > 0; off >>= 1) {
    snn  += __shfl_xor(snn,  off, 32);
    snn2 += __shfl_xor(snn2, off, 32);
    snd  += __shfl_xor(snd,  off, 32);
    snd2 += __shfl_xor(snd2, off, 32);
  }
  __shared__ float w0[8], w1[8], w2[8], w3[8];
  int warp = tid >> 5;
  if ((tid & 31) == 0) { w0[warp] = snn; w1[warp] = snn2; w2[warp] = snd; w3[warp] = snd2; }
  __syncthreads();
  if (tid == 0) {
    float a = 0.f, b = 0.f, c = 0.f, d = 0.f;
    #pragma unroll
    for (int i = 0; i < 8; ++i) { a += w0[i]; b += w1[i]; c += w2[i]; d += w3[i]; }
    const float cnt = 64.0f * 63.0f;                 // off-diagonal count
    float mu_nn  = a / cnt;
    float var_nn = (b - cnt * mu_nn * mu_nn) / (cnt - 1.0f);
    float sig_nn = sqrtf(fmaxf(var_nn, 0.0f));
    const float cntd = 4096.0f;
    float mu_nd  = c / cntd;
    float var_nd = (d - cntd * mu_nd * mu_nd) / (cntd - 1.0f);
    float sig_nd = sqrtf(fmaxf(var_nd, 0.0f));
    out[0] = -((mu_nn - mu_nd) / (sig_nn + sig_nd + 1e-8f));
  }
}

extern "C" void kernel_launch(void* const* d_in, const int* in_sizes, int n_in,
                              void* d_out, int out_size, void* d_ws, size_t ws_size,
                              hipStream_t stream) {
  const float* normal = (const float*)d_in[0];   // [64,256,64] f32
  const float* defect = (const float*)d_in[1];   // [64,256,64] f32
  char* ws = (char*)d_ws;
  const size_t emb = (size_t)N_IMG * P_PATCH * D_DIM;   // 1,048,576 elems
  bf16_t* a_bf   = (bf16_t*)ws;                         // 2 MB
  bf16_t* b_bf   = (bf16_t*)(ws + emb * sizeof(bf16_t));// 2 MB
  float*  sim_nn = (float*)(ws + emb * sizeof(bf16_t) * 2);
  float*  sim_nd = sim_nn + 4096;

  // 32768 rows total, 8 rows (waves) per block
  normalize_bf16_kernel<<<dim3(4096), dim3(256), 0, stream>>>(normal, defect, a_bf, b_bf);
  // one workgroup per image pair
  pair_sim_kernel<<<dim3(4096), dim3(512), 0, stream>>>(a_bf, a_bf, sim_nn);
  pair_sim_kernel<<<dim3(4096), dim3(512), 0, stream>>>(a_bf, b_bf, sim_nd);
  stats_kernel<<<dim3(1), dim3(256), 0, stream>>>(sim_nn, sim_nd, (float*)d_out);
}
